// MatteRefinementNetwork_4123168604889
// MI455X (gfx1250) — matrix operand order repaired and checked by reading, not compiled, not run on hardware
//
#include <hip/hip_runtime.h>
#include <hip/hip_bf16.h>

// ---------------------------------------------------------------------------
// MatteRefinementNetwork for MI455X (gfx1250, wave32, WMMA f16->f32)
// ---------------------------------------------------------------------------

typedef __attribute__((ext_vector_type(16))) _Float16 v16h;
typedef __attribute__((ext_vector_type(8)))  _Float16 v8h;
typedef __attribute__((ext_vector_type(2)))  _Float16 v2h;
typedef __attribute__((ext_vector_type(8)))  float    v8f;

#define KP       4096
#define HC_      512
#define WC_      512
#define HF_      2048
#define WF_      2048
#define WH_      1024          // half-res grid
#define NBINS    2048

// padded K (reduction) row lengths for f16 weight tables
#define KPAD1    576           // 9 kpos * 64 ch   (36 real -> pad 64)
#define KPAD2    288           // 9 kpos * 32 ch
#define KPAD3    864           // 9 kpos * 96 ch   (67 real -> pad 96)

// workspace byte offsets
#define HIST_OFF   0                       // 2*2048 ints = 16384 B
#define META_OFF   16384                   // 2*4 ints
#define CNT_OFF    16416                   // 2*2 ints
#define ZERO_BYTES 16448                   // memset range [0, ZERO_BYTES)
#define LIST_OFF   16448                   // 2*4096 ints = 32768 B
#define W1H_OFF    49216                   // 32*576 halves = 36864 B
#define W2H_OFF    86080                   // 64*288 halves = 36864 B
#define W3H_OFF    122944                  // 32*864 halves = 55296 B

__device__ __forceinline__ v16h combine16(v8h lo, v8h hi) {
    return __builtin_shufflevector(lo, hi, 0,1,2,3,4,5,6,7,8,9,10,11,12,13,14,15);
}
__device__ __forceinline__ v8f wmma_f16(v16h a, v16h b, v8f c) {
    // D = A(16x32 f16) x B(32x16 f16) + C(16x16 f32)
    return __builtin_amdgcn_wmma_f32_16x16x32_f16(false, a, false, b, (short)0, c, false, false);
}
__device__ __forceinline__ unsigned mono_key(float f) {
    unsigned u = __float_as_uint(f);
    return (u & 0x80000000u) ? ~u : (u | 0x80000000u);
}
__device__ __forceinline__ v2h pack2(float a, float b) {
    v2h r; r[0] = (_Float16)a; r[1] = (_Float16)b; return r;
}

// ---------------------------------------------------------------------------
// 1) weight prep: f32 -> f16, K-padded, row-major [N][Kpad], K = kpos*Cpad + c
// ---------------------------------------------------------------------------
__global__ void prep_w_kernel(const float* __restrict__ w1,
                              const float* __restrict__ w2,
                              const float* __restrict__ w3,
                              _Float16* __restrict__ w1h,
                              _Float16* __restrict__ w2h,
                              _Float16* __restrict__ w3h) {
    int t = blockIdx.x * 256 + threadIdx.x;
    if (t < 32 * KPAD1) {                    // w1: (32, 36, 3, 3)
        int n = t / KPAD1, kk = t % KPAD1, kpos = kk / 64, c = kk % 64;
        w1h[t] = (c < 36) ? (_Float16)w1[(n * 36 + c) * 9 + kpos] : (_Float16)0.f;
    }
    if (t < 64 * KPAD2) {                    // w2: (64, 32, 3, 3)
        int n = t / KPAD2, kk = t % KPAD2, kpos = kk / 32, c = kk % 32;
        w2h[t] = (_Float16)w2[(n * 32 + c) * 9 + kpos];
    }
    if (t < 32 * KPAD3) {                    // w3: (32, 67, 3, 3)
        int n = t / KPAD3, kk = t % KPAD3, kpos = kk / 96, c = kk % 96;
        w3h[t] = (c < 67) ? (_Float16)w3[(n * 67 + c) * 9 + kpos] : (_Float16)0.f;
    }
}

// ---------------------------------------------------------------------------
// 2) base image: out = up4(alpha)
// ---------------------------------------------------------------------------
__global__ __launch_bounds__(256) void up4_kernel(const float* __restrict__ alpha,
                                                  float* __restrict__ out) {
    int i = blockIdx.x * 256 + threadIdx.x;          // grid covers 2*2048*2048 exactly
    int b = i >> 22;                                 // / (2048*2048)
    int rem = i & (HF_ * WF_ - 1);
    int y = rem >> 11, x = rem & (WF_ - 1);
    out[i] = alpha[b * (HC_ * WC_) + (y >> 2) * WC_ + (x >> 2)];
}

// ---------------------------------------------------------------------------
// 3) top-K selection: histogram -> threshold -> compaction
// ---------------------------------------------------------------------------
__global__ __launch_bounds__(256) void hist_kernel(const float* __restrict__ err,
                                                   int* __restrict__ hist) {
    __shared__ int lh[NBINS];
    for (int t = threadIdx.x; t < NBINS; t += 256) lh[t] = 0;
    __syncthreads();
    int b = blockIdx.x >> 6;                 // 64 blocks per batch
    int blkInB = blockIdx.x & 63;
    const float* e = err + b * (HC_ * WC_);
    int base = blkInB * 4096;
    for (int t = threadIdx.x; t < 4096; t += 256)
        atomicAdd(&lh[mono_key(e[base + t]) >> 21], 1);
    __syncthreads();
    for (int t = threadIdx.x; t < NBINS; t += 256) {
        int v = lh[t];
        if (v) atomicAdd(&hist[b * NBINS + t], v);
    }
}

__global__ void thresh_kernel(const int* __restrict__ hist, int* __restrict__ meta) {
    if (threadIdx.x != 0) return;
    int b = blockIdx.x;
    const int* h = hist + b * NBINS;
    int accum = 0, T = 0, above = 0;
    for (int bin = NBINS - 1; bin >= 0; --bin) {
        int cnt = h[bin];
        if (accum + cnt >= KP) { T = bin; above = accum; break; }
        accum += cnt;
    }
    meta[b * 4 + 0] = T;
    meta[b * 4 + 1] = above;
    meta[b * 4 + 2] = KP - above;
}

__global__ __launch_bounds__(256) void compact_kernel(const float* __restrict__ err,
                                                      const int* __restrict__ meta,
                                                      int* __restrict__ counters,
                                                      int* __restrict__ list) {
    int b = blockIdx.x >> 10;                // 1024 blocks per batch
    int t = (blockIdx.x & 1023) * 256 + threadIdx.x;   // 0..262143
    int bin = (int)(mono_key(err[b * (HC_ * WC_) + t]) >> 21);
    int T = meta[b * 4 + 0], above = meta[b * 4 + 1], rem = meta[b * 4 + 2];
    if (bin > T) {
        int p = atomicAdd(&counters[b * 2 + 0], 1);
        list[b * KP + p] = t;
    } else if (bin == T) {
        int p = atomicAdd(&counters[b * 2 + 1], 1);
        if (p < rem) list[b * KP + above + p] = t;
    }
}

// ---------------------------------------------------------------------------
// 4) per-patch refinement net: one wave32 per patch, WMMA implicit-GEMM convs
// ---------------------------------------------------------------------------
__global__ __launch_bounds__(32) void patch_kernel(
    const float* __restrict__ alpha, const float* __restrict__ hidden,
    const float* __restrict__ input,
    const _Float16* __restrict__ w1h, const float* __restrict__ b1,
    const _Float16* __restrict__ w2h, const float* __restrict__ b2,
    const _Float16* __restrict__ w3h, const float* __restrict__ b3,
    const float* __restrict__ w4,     const float* __restrict__ b4,
    const int* __restrict__ list, float* __restrict__ out) {

    __shared__ alignas(16) _Float16 sP1[8 * 8 * 64];   // conv1 input, ch-minor (36->64)
    __shared__ alignas(16) _Float16 sP2[36 * 32];      // 6x6x32 (conv1 out / conv3 out)
    __shared__ alignas(16) _Float16 sP3[8 * 8 * 96];   // conv3 input (64 x2up + 3 mid ->96)
    __shared__ float sW4[288];

    const int blk  = blockIdx.x;
    const int b    = blk / KP;
    const int k    = blk - b * KP;
    const int lane = threadIdx.x;
    const int half = lane >> 4;          // 0 | 1 (lane group)
    const int col  = lane & 15;

    const int idx = list[b * KP + k];
    int r = idx >> 9, c = idx & 511;
    int r0 = 2 * r - 4; r0 = r0 < 0 ? 0 : (r0 > WH_ - 8 ? WH_ - 8 : r0);
    int c0 = 2 * c - 4; c0 = c0 < 0 ? 0 : (c0 > WH_ - 8 ? WH_ - 8 : c0);
    const int rf = 2 * r0, cf = 2 * c0;

    const float* alphaB  = alpha  + b * (HC_ * WC_);
    const float* hiddenB = hidden + b * (32 * HC_ * WC_);
    const float* inputB  = input  + b * (3 * HF_ * WF_);

    // ---- hoisted A-fragment bases (all inner-loop addressing is base + const)
    int pb1[3], pb3[3];
#pragma unroll
    for (int m = 0; m < 3; ++m) {
        int pos = m * 16 + col; if (pos > 35) pos = 35;   // pad rows (discarded)
        const int oy = pos / 6, ox = pos % 6;
        pb1[m] = (oy * 8 + ox) * 64 + half * 16;          // halves, 16B aligned
        pb3[m] = (oy * 8 + ox) * 96 + half * 16;
    }
    const int pb2 = ((col >> 2) * 6 + (col & 3)) * 32 + half * 16;

    // ---- stage 0: gather patch into sP1 [pix][ch] (packed 2-half stores)
    for (int px = lane; px < 64; px += 32) {
        int y = px >> 3, x = px & 7;
        int hy = (r0 + y) >> 1, hx = (c0 + x) >> 1;      // up2 source (512 grid)
        int fy = (r0 + y) << 1, fx = (c0 + x) << 1;      // ::2 downsample source
        const float* hB = hiddenB + hy * WC_ + hx;
        _Float16* dst = sP1 + px * 64;
        // ch0 = alpha, ch1..32 = hidden, ch33..35 = down_in, ch36..63 = 0
        *(v2h*)(dst + 0) = pack2(alphaB[hy * WC_ + hx], hB[0]);
#pragma unroll
        for (int j = 1; j < 16; ++j)
            *(v2h*)(dst + 2 * j) = pack2(hB[(2 * j - 1) * (HC_ * WC_)],
                                         hB[(2 * j) * (HC_ * WC_)]);
        *(v2h*)(dst + 32) = pack2(hB[31 * (HC_ * WC_)],
                                  inputB[0 * (HF_ * WF_) + fy * WF_ + fx]);
        *(v2h*)(dst + 34) = pack2(inputB[1 * (HF_ * WF_) + fy * WF_ + fx],
                                  inputB[2 * (HF_ * WF_) + fy * WF_ + fx]);
        v2h z2 = {}; v8h z8 = {};
        *(v2h*)(dst + 36) = z2; *(v2h*)(dst + 38) = z2;
        *(v8h*)(dst + 40) = z8; *(v8h*)(dst + 48) = z8; *(v8h*)(dst + 56) = z8;
    }
    for (int t = lane; t < 288; t += 32) sW4[t] = w4[t];
    __syncthreads();

    // ---- conv1: 8x8x36 -> 6x6x32, M=36(3 tiles), N=32(2 tiles), K=2 slices x 9 kpos
    const _Float16* w1b0 = w1h + col * KPAD1 + half * 16;
    const _Float16* w1b1 = w1h + (16 + col) * KPAD1 + half * 16;
    v8f a1[3][2];
#pragma unroll
    for (int m = 0; m < 3; ++m)
#pragma unroll
        for (int n = 0; n < 2; ++n) a1[m][n] = (v8f){};
#pragma unroll
    for (int kpos = 0; kpos < 9; ++kpos) {
        const int ko = ((kpos / 3) * 8 + (kpos % 3)) * 64;   // compile-time const
#pragma unroll
        for (int s = 0; s < 2; ++s) {
            v16h bf0 = *(const v16h*)(w1b0 + kpos * 64 + s * 32);
            v16h bf1 = *(const v16h*)(w1b1 + kpos * 64 + s * 32);
#pragma unroll
            for (int m = 0; m < 3; ++m) {
                const _Float16* p = sP1 + pb1[m] + ko + s * 32;
                v16h af = combine16(*(const v8h*)p, *(const v8h*)(p + 8));
                a1[m][0] = wmma_f16(af, bf0, a1[m][0]);
                a1[m][1] = wmma_f16(af, bf1, a1[m][1]);
            }
        }
    }
    // bias + relu -> sP2 [pos][ch]
#pragma unroll
    for (int m = 0; m < 3; ++m)
#pragma unroll
        for (int n = 0; n < 2; ++n) {
            const float bias = b1[n * 16 + col];
#pragma unroll
            for (int i = 0; i < 8; ++i) {
                int row = m * 16 + half * 8 + i;
                if (row < 36) {
                    float v = a1[m][n][i] + bias;
                    sP2[row * 32 + n * 16 + col] = (_Float16)(v > 0.f ? v : 0.f);
                }
            }
        }
    __syncthreads();

    // ---- conv2: 6x6x32 -> 4x4x64, M=16(1 tile), N=64(4 tiles), K=1 slice x 9 kpos
    const _Float16* w2b[4];
#pragma unroll
    for (int n = 0; n < 4; ++n) w2b[n] = w2h + (n * 16 + col) * KPAD2 + half * 16;
    v8f a2[4];
#pragma unroll
    for (int n = 0; n < 4; ++n) a2[n] = (v8f){};
#pragma unroll
    for (int kpos = 0; kpos < 9; ++kpos) {
        const int ko = ((kpos / 3) * 6 + (kpos % 3)) * 32;   // compile-time const
        const _Float16* p = sP2 + pb2 + ko;
        v16h af = combine16(*(const v8h*)p, *(const v8h*)(p + 8));
#pragma unroll
        for (int n = 0; n < 4; ++n) {
            v16h bf = *(const v16h*)(w2b[n] + kpos * 32);
            a2[n] = wmma_f16(af, bf, a2[n]);
        }
    }

    // ---- build sP3: zero, x2 up2 into ch 0..63, mid into ch 64..66
    {
        v8h z = {};
        for (int t = lane; t < (8 * 8 * 96) / 8; t += 32) ((v8h*)sP3)[t] = z;
    }
    __syncthreads();
#pragma unroll
    for (int n = 0; n < 4; ++n) {
        const float bias = b2[n * 16 + col];
        const int ch = n * 16 + col;
#pragma unroll
        for (int i = 0; i < 8; ++i) {
            int pos = half * 8 + i;                 // 0..15 over 4x4
            int oy = pos >> 2, ox = pos & 3;
            float v = a2[n][i] + bias; v = v > 0.f ? v : 0.f;
            _Float16 hv = (_Float16)v;
            int base = ((2 * oy) * 8 + 2 * ox) * 96 + ch;
            sP3[base]            = hv;
            sP3[base + 96]       = hv;
            sP3[base + 8 * 96]   = hv;
            sP3[base + 8 * 96 + 96] = hv;
        }
    }
    for (int t = lane; t < 192; t += 32) {          // 64 px * 3 ch (full-res gather)
        int px = t / 3, ch = t - px * 3;
        int y = px >> 3, x = px & 7;
        sP3[px * 96 + 64 + ch] =
            (_Float16)inputB[ch * (HF_ * WF_) + (rf + y) * WF_ + (cf + x)];
    }
    __syncthreads();

    // ---- conv3: 8x8x67 -> 6x6x32, M=36(3 tiles), N=32(2 tiles), K=3 slices x 9 kpos
    const _Float16* w3b0 = w3h + col * KPAD3 + half * 16;
    const _Float16* w3b1 = w3h + (16 + col) * KPAD3 + half * 16;
    v8f a3[3][2];
#pragma unroll
    for (int m = 0; m < 3; ++m)
#pragma unroll
        for (int n = 0; n < 2; ++n) a3[m][n] = (v8f){};
#pragma unroll
    for (int kpos = 0; kpos < 9; ++kpos) {
        const int ko = ((kpos / 3) * 8 + (kpos % 3)) * 96;   // compile-time const
#pragma unroll
        for (int s = 0; s < 3; ++s) {
            v16h bf0 = *(const v16h*)(w3b0 + kpos * 96 + s * 32);
            v16h bf1 = *(const v16h*)(w3b1 + kpos * 96 + s * 32);
#pragma unroll
            for (int m = 0; m < 3; ++m) {
                const _Float16* p = sP3 + pb3[m] + ko + s * 32;
                v16h af = combine16(*(const v8h*)p, *(const v8h*)(p + 8));
                a3[m][0] = wmma_f16(af, bf0, a3[m][0]);
                a3[m][1] = wmma_f16(af, bf1, a3[m][1]);
            }
        }
    }
    __syncthreads();                                 // sP2 reuse
#pragma unroll
    for (int m = 0; m < 3; ++m)
#pragma unroll
        for (int n = 0; n < 2; ++n) {
            const float bias = b3[n * 16 + col];
#pragma unroll
            for (int i = 0; i < 8; ++i) {
                int row = m * 16 + half * 8 + i;
                if (row < 36) {
                    float v = a3[m][n][i] + bias;
                    sP2[row * 32 + n * 16 + col] = (_Float16)(v > 0.f ? v : 0.f);
                }
            }
        }
    __syncthreads();

    // ---- conv4 (32->1, only 16 outputs): scalar, then scatter 4x4 into out
    if (lane < 16) {
        const int oy = lane >> 2, ox = lane & 3;
        float sum = b4[0];
#pragma unroll
        for (int c2 = 0; c2 < 32; ++c2)
#pragma unroll
            for (int ky = 0; ky < 3; ++ky)
#pragma unroll
                for (int kx = 0; kx < 3; ++kx)
                    sum += (float)sP2[((oy + ky) * 6 + ox + kx) * 32 + c2]
                         * sW4[(c2 * 3 + ky) * 3 + kx];
        out[b * (HF_ * WF_) + (rf + 2 + oy) * WF_ + (cf + 2 + ox)] = sum;
    }
}

// ---------------------------------------------------------------------------
// launch
// ---------------------------------------------------------------------------
extern "C" void kernel_launch(void* const* d_in, const int* in_sizes, int n_in,
                              void* d_out, int out_size, void* d_ws, size_t ws_size,
                              hipStream_t stream) {
    const float* alpha  = (const float*)d_in[0];
    const float* err    = (const float*)d_in[1];
    const float* hidden = (const float*)d_in[2];
    const float* input  = (const float*)d_in[3];
    const float* w1 = (const float*)d_in[4];
    const float* b1 = (const float*)d_in[5];
    const float* w2 = (const float*)d_in[6];
    const float* b2 = (const float*)d_in[7];
    const float* w3 = (const float*)d_in[8];
    const float* b3 = (const float*)d_in[9];
    const float* w4 = (const float*)d_in[10];
    const float* b4 = (const float*)d_in[11];
    float* out = (float*)d_out;

    char* ws = (char*)d_ws;
    int* hist      = (int*)(ws + HIST_OFF);
    int* meta      = (int*)(ws + META_OFF);
    int* counters  = (int*)(ws + CNT_OFF);
    int* list      = (int*)(ws + LIST_OFF);
    _Float16* w1h  = (_Float16*)(ws + W1H_OFF);
    _Float16* w2h  = (_Float16*)(ws + W2H_OFF);
    _Float16* w3h  = (_Float16*)(ws + W3H_OFF);

    hipMemsetAsync(ws, 0, ZERO_BYTES, stream);

    prep_w_kernel<<<(32 * KPAD3 + 255) / 256, 256, 0, stream>>>(w1, w2, w3, w1h, w2h, w3h);
    up4_kernel<<<(2 * HF_ * WF_) / 256, 256, 0, stream>>>(alpha, out);
    hist_kernel<<<128, 256, 0, stream>>>(err, hist);
    thresh_kernel<<<2, 32, 0, stream>>>(hist, meta);
    compact_kernel<<<2048, 256, 0, stream>>>(err, meta, counters, list);
    patch_kernel<<<2 * KP, 32, 0, stream>>>(alpha, hidden, input,
                                            w1h, b1, w2h, b2, w3h, b3, w4, b4,
                                            list, out);
}
